// Full_Tmix_Params_6562710028360
// MI455X (gfx1250) — matrix operand order repaired and checked by reading, not compiled, not run on hardware
//
#include <hip/hip_runtime.h>
#include <hip/hip_bf16.h>

// ---------------------------------------------------------------------------
// RWKV-7 time-mix for MI455X (gfx1250, wave32).
// Compute-bound part = 4 square + 8 LoRA GEMMs -> bf16 WMMA (v_wmma_f32_16x16x32_bf16),
// f32 accumulate. Elementwise parts = wave-per-head kernels (shfl reductions).
// ---------------------------------------------------------------------------

typedef __attribute__((ext_vector_type(16))) __bf16 v16bf;
typedef __attribute__((ext_vector_type(8)))  __bf16 v8bf;
typedef __attribute__((ext_vector_type(8)))  float  v8f;

#define B_DIM  2
#define T_DIM  4096
#define C_DIM  2048
#define M_DIM  (B_DIM * T_DIM)   // 8192 tokens
#define H_DIM  32
#define HS_DIM 64

// ---------------------------------------------------------------------------
// Generic NT GEMM:  C[M,N](f32) = A[M,K](bf16, lda) * Bt[N,K](bf16, ldb)^T
// Block: 256 thr = 8 waves (2 M x 4 N). Block tile 128x128, wave tile 64x32
// (4x2 WMMA -> 8 v_wmma per K-step, 1.5 ds_load_b128 per WMMA).
// K stepped by 32 through LDS. M%128==0, N%128==0, K%32==0 required.
// ---------------------------------------------------------------------------
__global__ __launch_bounds__(256)
void gemm_bf16_nt(const __bf16* __restrict__ A, int lda,
                  const __bf16* __restrict__ Bt, int ldb,
                  float* __restrict__ Cm, int ldc, int K)
{
    __shared__ __align__(64) __bf16 As[128 * 32];
    __shared__ __align__(64) __bf16 Bs[128 * 32];

    const int tid   = threadIdx.x;
    const int wave  = tid >> 5;
    const int lane  = tid & 31;
    const int wm    = (wave & 1) * 64;      // wave M offset in block tile
    const int wn    = (wave >> 1) * 32;     // wave N offset in block tile
    const long bm   = (long)blockIdx.y * 128;
    const long bn   = (long)blockIdx.x * 128;
    const int row16 = lane & 15;
    const int hi    = lane >> 4;

    v8f acc[4][2];
    #pragma unroll
    for (int mi = 0; mi < 4; ++mi)
        #pragma unroll
        for (int ni = 0; ni < 2; ++ni)
            acc[mi][ni] = (v8f){};

    // cooperative tile-load assignments: 128 rows x 32 cols, 16 halves/thread
    const int lr = tid >> 1;            // 0..127 (tile row)
    const int lc = (tid & 1) * 16;      // 0/16   (tile col, 16 halves = 32B)

    for (int k0 = 0; k0 < K; k0 += 32) {
        // stage A tile (128x32 bf16)
        {
            const uint4* src = (const uint4*)(A + (bm + lr) * (long)lda + k0 + lc);
            uint4 q0 = src[0];
            uint4 q1 = src[1];
            *(uint4*)(&As[lr * 32 + lc])     = q0;
            *(uint4*)(&As[lr * 32 + lc + 8]) = q1;
        }
        // stage B tile (128x32 bf16)
        {
            const uint4* src = (const uint4*)(Bt + (bn + lr) * (long)ldb + k0 + lc);
            uint4 q0 = src[0];
            uint4 q1 = src[1];
            *(uint4*)(&Bs[lr * 32 + lc])     = q0;
            *(uint4*)(&Bs[lr * 32 + lc + 8]) = q1;
        }
        __syncthreads();

        // A fragments: lane (row16,hi) holds K = hi*16 + 0..15 (contiguous, matches B)
        v16bf af[4];
        #pragma unroll
        for (int mi = 0; mi < 4; ++mi) {
            const int r = wm + mi * 16 + row16;
            v8bf lo = *(const v8bf*)&As[r * 32 + hi * 16];
            v8bf hv = *(const v8bf*)&As[r * 32 + hi * 16 + 8];
            af[mi] = __builtin_shufflevector(lo, hv,
                     0,1,2,3,4,5,6,7,8,9,10,11,12,13,14,15);
        }
        // B fragments: lane (col=row16, hi) holds K = hi*16 + 0..15 (contiguous)
        v16bf bf[2];
        #pragma unroll
        for (int ni = 0; ni < 2; ++ni)
            bf[ni] = *(const v16bf*)&Bs[(wn + ni * 16 + row16) * 32 + hi * 16];

        #pragma unroll
        for (int mi = 0; mi < 4; ++mi)
            #pragma unroll
            for (int ni = 0; ni < 2; ++ni)
                acc[mi][ni] = __builtin_amdgcn_wmma_f32_16x16x32_bf16(
                    false, af[mi], false, bf[ni], (short)0, acc[mi][ni],
                    false, false);
        __syncthreads();
    }

    // C/D layout: VGPR i -> row hi*8+i; col = lane&15
    #pragma unroll
    for (int mi = 0; mi < 4; ++mi) {
        #pragma unroll
        for (int ni = 0; ni < 2; ++ni) {
            const long r = bm + wm + mi * 16 + hi * 8;
            const long c = bn + wn + ni * 16 + row16;
            #pragma unroll
            for (int i = 0; i < 8; ++i)
                Cm[(r + i) * ldc + c] = acc[mi][ni][i];
        }
    }
}

// ---------------------------------------------------------------------------
// Token-shift mix: out = bf16( x + ((t>0 ? x[t-1] : 0) - x) * coef[c] )
// ---------------------------------------------------------------------------
__global__ __launch_bounds__(256)
void mix_kernel(const float* __restrict__ x, const float* __restrict__ coef,
                __bf16* __restrict__ out)
{
    const long idx = (long)blockIdx.x * 256 + threadIdx.x;   // over M*C, exact
    const long m   = idx >> 11;                              // /C (2048)
    const int  c   = (int)(idx & (C_DIM - 1));
    const long t   = m & (T_DIM - 1);                        // m % T
    const float xm = x[idx];
    const float xp = (t > 0) ? x[idx - C_DIM] : 0.0f;
    out[idx] = (__bf16)(xm + (xp - xm) * coef[c]);
}

// f32 -> bf16 copy (square weights already in [out,in] = NT layout)
__global__ __launch_bounds__(256)
void cvt_kernel(const float* __restrict__ in, __bf16* __restrict__ out)
{
    const long idx = (long)blockIdx.x * 256 + threadIdx.x;
    out[idx] = (__bf16)in[idx];
}

// transpose+convert: in[Kd,Nd] f32 row-major -> out[Npad,Kd] bf16, zero pad rows
__global__ __launch_bounds__(256)
void tcvt_kernel(const float* __restrict__ in, __bf16* __restrict__ out,
                 int Kd, int Nd, int Npad)
{
    const long idx = (long)blockIdx.x * 256 + threadIdx.x;
    if (idx >= (long)Npad * Kd) return;
    const int n = (int)(idx / Kd);
    const int k = (int)(idx % Kd);
    out[idx] = (n < Nd) ? (__bf16)in[(long)k * Nd + n] : (__bf16)0.0f;
}

// LoRA mid activations over h[M,640]: [0,128)=tanh(hw), [128,384)=identity
// (ha, hv), [384,640)=sigmoid(hg). Emit bf16 for stage-2 GEMMs.
__global__ __launch_bounds__(256)
void mid_kernel(const float* __restrict__ h, __bf16* __restrict__ th)
{
    const long idx = (long)blockIdx.x * 256 + threadIdx.x;   // over M*640, exact
    const int col  = (int)(idx % 640);
    float v = h[idx];
    if (col < 128)        v = tanhf(v);
    else if (col >= 384)  v = 1.0f / (1.0f + expf(-v));
    th[idx] = (__bf16)v;
}

__device__ __forceinline__ float wsum(float v)
{
    #pragma unroll
    for (int m = 16; m > 0; m >>= 1) v += __shfl_xor(v, m, 32);
    return v;
}

// ---------------------------------------------------------------------------
// Fused post kernel. One wave32 = one (token, head) of 64 channels (2/lane).
// Computes w, a, kk (in-place over staged pre-activations in d_out segments),
// value mixing with v_first, groupnorm(y), per-head dot, gate, and bf16 z.
// ---------------------------------------------------------------------------
__global__ __launch_bounds__(256)
void post_kernel(const float* __restrict__ rbuf, float* __restrict__ kseg,
                 const float* __restrict__ vbuf, const float* __restrict__ vfirst,
                 const float* __restrict__ y,
                 float* __restrict__ wseg, float* __restrict__ aseg,
                 const float* __restrict__ vpre, const float* __restrict__ gpre,
                 const float* __restrict__ w0, const float* __restrict__ a0,
                 const float* __restrict__ v0,
                 const float* __restrict__ k_k, const float* __restrict__ k_a,
                 const float* __restrict__ r_k,
                 const float* __restrict__ ln_g, const float* __restrict__ ln_b,
                 __bf16* __restrict__ z)
{
    const int  wave = threadIdx.x >> 5;
    const int  lane = threadIdx.x & 31;
    const long gh   = (long)blockIdx.x * 8 + wave;   // token*H + head
    const long m    = gh >> 5;                        // /H (32)
    const int  h    = (int)(gh & 31);
    const int  c    = h * HS_DIM + lane * 2;          // channel (also r_k index)
    const long base = m * C_DIM + c;

    const float r0 = rbuf[base],  r1 = rbuf[base + 1];
    const float kr0 = kseg[base], kr1 = kseg[base + 1];
    const float vr0 = vbuf[base], vr1 = vbuf[base + 1];
    const float y0v = y[base],    y1v = y[base + 1];

    // decay: w = -softplus(-(w0 + pre)) - 0.5
    const float wv0 = -log1pf(expf(-(w0[c] + wseg[base]))) - 0.5f;
    const float wv1 = -log1pf(expf(-(w0[c + 1] + wseg[base + 1]))) - 0.5f;
    // in-context rate: a = sigmoid(a0 + pre)
    const float av0 = 1.0f / (1.0f + expf(-(a0[c] + aseg[base])));
    const float av1 = 1.0f / (1.0f + expf(-(a0[c + 1] + aseg[base + 1])));
    // value mix with v_first
    const float s0 = 1.0f / (1.0f + expf(-(v0[c] + vpre[base])));
    const float s1 = 1.0f / (1.0f + expf(-(v0[c + 1] + vpre[base + 1])));
    const float vv0 = vr0 + (vfirst[base] - vr0) * s0;
    const float vv1 = vr1 + (vfirst[base + 1] - vr1) * s1;
    // kk = head-normalize(k * k_k)
    float kk0 = kr0 * k_k[c];
    float kk1 = kr1 * k_k[c + 1];
    const float ss   = wsum(kk0 * kk0 + kk1 * kk1);
    const float invn = 1.0f / fmaxf(sqrtf(ss), 1e-12f);
    kk0 *= invn; kk1 *= invn;
    // k scaled by (1 + (a-1)*k_a)
    const float kn0 = kr0 * (1.0f + (av0 - 1.0f) * k_a[c]);
    const float kn1 = kr1 * (1.0f + (av1 - 1.0f) * k_a[c + 1]);
    // groupnorm(y) per head + affine
    const float sy  = wsum(y0v + y1v);
    const float sy2 = wsum(y0v * y0v + y1v * y1v);
    const float mu  = sy * (1.0f / 64.0f);
    const float var = sy2 * (1.0f / 64.0f) - mu * mu;
    const float istd = rsqrtf(var + 0.00064f);
    const float yn0 = (y0v - mu) * istd * ln_g[c] + ln_b[c];
    const float yn1 = (y1v - mu) * istd * ln_g[c + 1] + ln_b[c + 1];
    // per-head dot(r, k, r_k)
    const float dot = wsum(r0 * kn0 * r_k[c] + r1 * kn1 * r_k[c + 1]);
    const float y20 = yn0 + dot * vv0;
    const float y21 = yn1 + dot * vv1;
    const float g0v = gpre[base], g1v = gpre[base + 1];

    wseg[base] = wv0;  wseg[base + 1] = wv1;
    aseg[base] = av0;  aseg[base + 1] = av1;
    kseg[base] = kk0;  kseg[base + 1] = kk1;
    z[base]     = (__bf16)(y20 * g0v);
    z[base + 1] = (__bf16)(y21 * g1v);
}

// ---------------------------------------------------------------------------
extern "C" void kernel_launch(void* const* d_in, const int* in_sizes, int n_in,
                              void* d_out, int out_size, void* d_ws, size_t ws_size,
                              hipStream_t stream)
{
    (void)in_sizes; (void)n_in; (void)out_size; (void)ws_size;

    const float* x      = (const float*)d_in[0];
    const float* vfirst = (const float*)d_in[1];
    const float* y      = (const float*)d_in[2];
    const float* x_r = (const float*)d_in[3];
    const float* x_w = (const float*)d_in[4];
    const float* x_k = (const float*)d_in[5];
    const float* x_v = (const float*)d_in[6];
    const float* x_a = (const float*)d_in[7];
    const float* x_g = (const float*)d_in[8];
    const float* w0 = (const float*)d_in[9];
    const float* w1 = (const float*)d_in[10];
    const float* w2 = (const float*)d_in[11];
    const float* a0 = (const float*)d_in[12];
    const float* a1 = (const float*)d_in[13];
    const float* a2 = (const float*)d_in[14];
    const float* v0 = (const float*)d_in[15];
    const float* v1 = (const float*)d_in[16];
    const float* v2 = (const float*)d_in[17];
    const float* g1 = (const float*)d_in[18];
    const float* g2 = (const float*)d_in[19];
    const float* k_k = (const float*)d_in[20];
    const float* k_a = (const float*)d_in[21];
    const float* r_k = (const float*)d_in[22];
    const float* W_r = (const float*)d_in[23];
    const float* W_k = (const float*)d_in[24];
    const float* W_v = (const float*)d_in[25];
    const float* W_o = (const float*)d_in[26];
    const float* ln_g = (const float*)d_in[27];
    const float* ln_b = (const float*)d_in[28];

    const long MC = (long)M_DIM * C_DIM;     // 16,777,216
    float* outp = (float*)d_out;             // out  [M,C]
    float* wseg = outp + MC;                 // w    [M,C] (staged w_pre, then w)
    float* aseg = outp + 2 * MC;             // a    [M,C] (staged a_pre, then a)
    float* kseg = outp + 3 * MC;             // kk   [M,C] (staged k,    then kk)

    // workspace carve-out (256B aligned chunks)
    char* cur = (char*)d_ws;
    auto alloc = [&](size_t bytes) -> void* {
        void* p = (void*)cur;
        cur += (bytes + 255) & ~(size_t)255;
        return p;
    };
    const size_t WSQ = (size_t)C_DIM * C_DIM * sizeof(__bf16);
    __bf16* Wr_bf = (__bf16*)alloc(WSQ);
    __bf16* Wk_bf = (__bf16*)alloc(WSQ);
    __bf16* Wv_bf = (__bf16*)alloc(WSQ);
    __bf16* Wo_bf = (__bf16*)alloc(WSQ);
    __bf16* w1t = (__bf16*)alloc((size_t)128 * C_DIM * 2);  // [128,2048]
    __bf16* w2t = (__bf16*)alloc((size_t)C_DIM * 128 * 2);  // [2048,128]
    __bf16* a1t = (__bf16*)alloc((size_t)128 * C_DIM * 2);  // [128,2048]
    __bf16* a2t = (__bf16*)alloc((size_t)C_DIM * 128 * 2);  // [2048,128]
    __bf16* v1t = (__bf16*)alloc((size_t)128 * C_DIM * 2);  // [64->128 pad, 2048]
    __bf16* v2t = (__bf16*)alloc((size_t)C_DIM * 64 * 2);   // [2048,64]
    __bf16* g1t = (__bf16*)alloc((size_t)256 * C_DIM * 2);  // [224->256 pad, 2048]
    __bf16* g2t = (__bf16*)alloc((size_t)C_DIM * 224 * 2);  // [2048,224]
    __bf16* mixbuf = (__bf16*)alloc((size_t)MC * 2);        // reused per branch
    float*  rbuf   = (float*)alloc((size_t)MC * 4);
    float*  vbuf   = (float*)alloc((size_t)MC * 4);
    float*  hbuf   = (float*)alloc((size_t)M_DIM * 640 * 4); // [hw|ha|hv|hg]
    __bf16* thbuf  = (__bf16*)alloc((size_t)M_DIM * 640 * 2);
    float*  vpbuf  = (float*)alloc((size_t)MC * 4);
    float*  gbuf   = (float*)alloc((size_t)MC * 4);
    __bf16* zbuf   = (__bf16*)alloc((size_t)MC * 2);

    const int TPB = 256;
    const long SQ = (long)C_DIM * C_DIM;

    // weight conversions (NT layout for x @ W.T is W itself)
    cvt_kernel<<<SQ / TPB, TPB, 0, stream>>>(W_r, Wr_bf);
    cvt_kernel<<<SQ / TPB, TPB, 0, stream>>>(W_k, Wk_bf);
    cvt_kernel<<<SQ / TPB, TPB, 0, stream>>>(W_v, Wv_bf);
    cvt_kernel<<<SQ / TPB, TPB, 0, stream>>>(W_o, Wo_bf);
    auto tc = [&](const float* in, __bf16* out, int Kd, int Nd, int Npad) {
        const long n = (long)Npad * Kd;
        tcvt_kernel<<<(n + TPB - 1) / TPB, TPB, 0, stream>>>(in, out, Kd, Nd, Npad);
    };
    tc(w1, w1t, C_DIM, 128, 128);
    tc(w2, w2t, 128, C_DIM, C_DIM);
    tc(a1, a1t, C_DIM, 128, 128);
    tc(a2, a2t, 128, C_DIM, C_DIM);
    tc(v1, v1t, C_DIM, 64, 128);
    tc(v2, v2t, 64, C_DIM, C_DIM);
    tc(g1, g1t, C_DIM, 224, 256);
    tc(g2, g2t, 224, C_DIM, C_DIM);

    auto gemm = [&](const __bf16* A, int lda, const __bf16* Bt, int ldb,
                    float* Cm, int ldc, int N, int K) {
        dim3 grid(N / 128, M_DIM / 128);
        gemm_bf16_nt<<<grid, TPB, 0, stream>>>(A, lda, Bt, ldb, Cm, ldc, K);
    };
    auto mix = [&](const float* coef) {
        mix_kernel<<<MC / TPB, TPB, 0, stream>>>(x, coef, mixbuf);
    };

    // ---- stage 1 projections (mix buffer reused serially) ----
    mix(x_r); gemm(mixbuf, C_DIM, Wr_bf, C_DIM, rbuf, C_DIM, C_DIM, C_DIM);
    mix(x_w); gemm(mixbuf, C_DIM, w1t, C_DIM, hbuf + 0,   640, 128, C_DIM);
    mix(x_k); gemm(mixbuf, C_DIM, Wk_bf, C_DIM, kseg, C_DIM, C_DIM, C_DIM);
    mix(x_v); gemm(mixbuf, C_DIM, Wv_bf, C_DIM, vbuf, C_DIM, C_DIM, C_DIM);
              gemm(mixbuf, C_DIM, v1t, C_DIM, hbuf + 256, 640, 128, C_DIM);
    mix(x_a); gemm(mixbuf, C_DIM, a1t, C_DIM, hbuf + 128, 640, 128, C_DIM);
    mix(x_g); gemm(mixbuf, C_DIM, g1t, C_DIM, hbuf + 384, 640, 256, C_DIM);

    // ---- LoRA mid activations ----
    mid_kernel<<<((long)M_DIM * 640) / TPB, TPB, 0, stream>>>(hbuf, thbuf);

    // ---- stage 2 LoRA GEMMs ----
    gemm(thbuf + 0,   640, w2t, 128, wseg,  C_DIM, C_DIM, 128);  // w_pre
    gemm(thbuf + 128, 640, a2t, 128, aseg,  C_DIM, C_DIM, 128);  // a_pre
    gemm(thbuf + 256, 640, v2t, 64,  vpbuf, C_DIM, C_DIM, 64);   // v-mix pre
    gemm(thbuf + 384, 640, g2t, 224, gbuf,  C_DIM, C_DIM, 224);  // g (final)

    // ---- fused post (w/a/kk finalize in-place, groupnorm, dot, gate, z) ----
    post_kernel<<<(M_DIM * H_DIM) / 8, TPB, 0, stream>>>(
        rbuf, kseg, vbuf, vfirst, y, wseg, aseg, vpbuf, gbuf,
        w0, a0, v0, k_k, k_a, r_k, ln_g, ln_b, zbuf);

    // ---- output projection ----
    gemm(zbuf, C_DIM, Wo_bf, C_DIM, outp, C_DIM, C_DIM, C_DIM);
}
